// BitSet_43078521979160
// MI455X (gfx1250) — 8-bit matrix operand order repaired and checked by reading, compiled, not run on hardware
//
#include <hip/hip_runtime.h>

typedef __attribute__((ext_vector_type(8))) int v8i;

constexpr int BATCH       = 8192;
constexpr int NUM_REFS    = 2048;
constexpr int DIM         = 256;
constexpr int M_TILE      = 128;
constexpr int N_CHUNK     = 16;
constexpr int REF_SPLIT   = 4;
constexpr int REFS_PER_WG = NUM_REFS / REF_SPLIT;
constexpr int NB_ITERS    = REFS_PER_WG / N_CHUNK;
constexpr int THREADS     = 256;

__global__ __launch_bounds__(256) void init_kernel(int* __restrict__ out_i) {
  int i = blockIdx.x * blockDim.x + threadIdx.x;
  if (i < BATCH) out_i[i] = 0x7FFFFFFF;
}

__global__ __launch_bounds__(256) void convertR_kernel(
    const float* __restrict__ R, unsigned char* __restrict__ B8) {
  const int t = blockIdx.x * blockDim.x + threadIdx.x;
  const float* src = R + (size_t)t * 16;
  unsigned wv[4];
#pragma unroll
  for (int q = 0; q < 4; ++q) {
    float4 f = *(const float4*)(src + q * 4);
    wv[q] = (unsigned)(int)f.x | ((unsigned)(int)f.y << 8)
          | ((unsigned)(int)f.z << 16) | ((unsigned)(int)f.w << 24);
  }
  uint4 w; w.x = wv[0]; w.y = wv[1]; w.z = wv[2]; w.w = wv[3];
  *(uint4*)(B8 + (size_t)t * 16) = w;
}

__global__ __launch_bounds__(THREADS) void hamming_min_kernel(
    const float* __restrict__ states, const unsigned char* __restrict__ B8,
    int* __restrict__ out_i) {
  __shared__ __align__(16) signed char   sA[M_TILE * DIM];
  __shared__ __align__(16) unsigned char sB[2][N_CHUNK * DIM];
  __shared__ int popLds[THREADS];

  const int tid       = threadIdx.x;
  const int lane      = tid & 31;
  const int wave      = tid >> 5;
  const int blockRow0 = blockIdx.x * M_TILE;
  const int refBase0  = blockIdx.y * REFS_PER_WG;

  auto issue_copy = [&](int chunk, int buf) {
    const unsigned char* g =
        B8 + (size_t)(refBase0 + chunk * N_CHUNK) * DIM + tid * 16;
    unsigned lds = (unsigned)(size_t)(&sB[buf][tid * 16]);
    unsigned long long ga = (unsigned long long)(size_t)g;
    asm volatile("global_load_async_to_lds_b128 %0, %1, off"
                 :: "v"(lds), "v"(ga) : "memory");
  };

  issue_copy(0, 0);
  issue_copy(1, 1);

  {
    const int row     = tid >> 1;
    const int colBase = (tid & 1) * 128;
    const float* src  = states + (size_t)(blockRow0 + row) * DIM + colBase;
    int pop = 0;
#pragma unroll
    for (int c = 0; c < 128; c += 16) {
      unsigned wv[4];
#pragma unroll
      for (int q = 0; q < 4; ++q) {
        float4 f = *(const float4*)(src + c + q * 4);
        int s0 = (int)f.x, s1 = (int)f.y, s2 = (int)f.z, s3 = (int)f.w;
        pop += s0 + s1 + s2 + s3;
        wv[q] = (unsigned)((1 - 2 * s0) & 0xFF)
              | ((unsigned)((1 - 2 * s1) & 0xFF) << 8)
              | ((unsigned)((1 - 2 * s2) & 0xFF) << 16)
              | ((unsigned)((1 - 2 * s3) & 0xFF) << 24);
      }
      uint4 w; w.x = wv[0]; w.y = wv[1]; w.z = wv[2]; w.w = wv[3];
      *(uint4*)(sA + row * DIM + colBase + c) = w;
    }
    popLds[tid] = pop;
  }
  __syncthreads();

  const int halfSel = (lane >> 4) & 1;
  const int mrow    = lane & 15;
  const int aRow    = wave * 16 + mrow;
  v8i afrag[4];
#pragma unroll
  for (int kc = 0; kc < 4; ++kc) {
    const signed char* base = sA + aRow * DIM + kc * 64 + halfSel * 16;
#pragma unroll
    for (int j = 0; j < 4; ++j) {
      uint2 d = *(const uint2*)(base + (j & 1) * 8 + (j >> 1) * 32);
      afrag[kc][2 * j]     = (int)d.x;
      afrag[kc][2 * j + 1] = (int)d.y;
    }
  }

  v8i minD;
#pragma unroll
  for (int v = 0; v < 8; ++v) minD[v] = 0x7FFFFFFF;

  const int bCol = lane & 15;

  for (int nb = 0; nb < NB_ITERS; ++nb) {
    if (nb + 1 < NB_ITERS) asm volatile("s_wait_asynccnt 1" ::: "memory");
    else                   asm volatile("s_wait_asynccnt 0" ::: "memory");
    __syncthreads();

    const unsigned char* bbuf = sB[nb & 1];
    v8i c = {};
#pragma unroll
    for (int kc = 0; kc < 4; ++kc) {
      const unsigned char* base = bbuf + bCol * DIM + kc * 64 + halfSel * 16;
      uint4 lo = *(const uint4*)(base);
      uint4 hi = *(const uint4*)(base + 32);
      v8i b;
      b[0] = (int)lo.x; b[1] = (int)lo.y; b[2] = (int)lo.z; b[3] = (int)lo.w;
      b[4] = (int)hi.x; b[5] = (int)hi.y; b[6] = (int)hi.z; b[7] = (int)hi.w;
      c = __builtin_amdgcn_wmma_i32_16x16x64_iu8(
           true, afrag[kc],  false, b, c,
           false,  false);
    }
#pragma unroll
    for (int v = 0; v < 8; ++v) minD[v] = min(minD[v], c[v]);
    __syncthreads();

    if (nb + 2 < NB_ITERS) issue_copy(nb + 2, nb & 1);
  }

#pragma unroll
  for (int v = 0; v < 8; ++v) {
    int val = minD[v];
    val = min(val, __shfl_xor(val, 1, 32));
    val = min(val, __shfl_xor(val, 2, 32));
    val = min(val, __shfl_xor(val, 4, 32));
    val = min(val, __shfl_xor(val, 8, 32));
    minD[v] = val;
  }
  if ((lane & 15) == 0) {
    const int rowOff = wave * 16 + halfSel * 8;
#pragma unroll
    for (int v = 0; v < 8; ++v) {
      const int rowInBlock = rowOff + v;
      const int pop = popLds[2 * rowInBlock] + popLds[2 * rowInBlock + 1];
      atomicMin(out_i + blockRow0 + rowInBlock, minD[v] + pop);
    }
  }
}

__global__ __launch_bounds__(256) void finalize_kernel(int* __restrict__ buf) {
  int i = blockIdx.x * blockDim.x + threadIdx.x;
  if (i < BATCH) {
    int v = buf[i];
    ((float*)buf)[i] = (float)v;
  }
}

extern "C" void kernel_launch(void* const* d_in, const int* in_sizes, int n_in,
                              void* d_out, int out_size, void* d_ws, size_t ws_size,
                              hipStream_t stream) {
  (void)in_sizes; (void)n_in; (void)out_size; (void)ws_size;
  const float* states = (const float*)d_in[0];
  const float* R      = (const float*)d_in[1];
  int* out_i          = (int*)d_out;
  unsigned char* B8   = (unsigned char*)d_ws;

  init_kernel<<<BATCH / 256, 256, 0, stream>>>(out_i);
  convertR_kernel<<<(NUM_REFS * DIM / 16) / 256, 256, 0, stream>>>(R, B8);
  dim3 grid(BATCH / M_TILE, REF_SPLIT);
  hamming_min_kernel<<<grid, THREADS, 0, stream>>>(states, B8, out_i);
  finalize_kernel<<<BATCH / 256, 256, 0, stream>>>(out_i);
}
